// Attention_74363063763568
// MI455X (gfx1250) — matrix operand order repaired and checked by reading, not compiled, not run on hardware
//
#include <hip/hip_runtime.h>
#include <hip/hip_bf16.h>

typedef __bf16 bf16_t;
typedef __attribute__((ext_vector_type(16))) __bf16 v16bf;
typedef __attribute__((ext_vector_type(8)))  __bf16 v8bf;
typedef __attribute__((ext_vector_type(8)))  float  v8f;
typedef __attribute__((ext_vector_type(4)))  unsigned int u32x4;
typedef __attribute__((ext_vector_type(8)))  int i32x8;
typedef __attribute__((ext_vector_type(4)))  int i32x4;

#define DIM     768
#define NHEADS  12
#define HDIM    64
#define NTOK    577
#define NPAD    640           // 10 key-tiles of 64
#define BATCH   32
#define MROWS   (BATCH * NTOK)   // 18464
#define NN      (NTOK * NTOK)    // 332929

static __device__ __forceinline__ int lane_id() { return (int)(threadIdx.x & 31); }

// ---- WMMA fragment helpers (ISA 7.12.2 layouts, wave32) --------------------
// A 16x32 bf16: lane = half*16 + row; elems 0..7 -> K = half*8 + e,
//                                     elems 8..15 -> K = 16 + half*8 + e.
static __device__ __forceinline__ v16bf load_frag_a(const bf16_t* p, int ld) {
  const int l = lane_id();
  const bf16_t* r = p + (l & 15) * ld + (l >> 4) * 8;
  v8bf lo = *reinterpret_cast<const v8bf*>(r);
  v8bf hi = *reinterpret_cast<const v8bf*>(r + 16);
  return __builtin_shufflevector(lo, hi, 0, 1, 2, 3, 4, 5, 6, 7,
                                         8, 9, 10, 11, 12, 13, 14, 15);
}

// B 32x16 bf16: lane = half*16 + col; elems 0..7 -> K = half*8 + e,
//                                     elems 8..15 -> K = 16 + half*8 + e.
// p points at tile stored [col][K] (K contiguous), ld = K-stride.
static __device__ __forceinline__ v16bf load_frag_b(const bf16_t* p, int ld) {
  const int l = lane_id();
  const bf16_t* r = p + (l & 15) * ld + (l >> 4) * 8;
  v8bf lo = *reinterpret_cast<const v8bf*>(r);
  v8bf hi = *reinterpret_cast<const v8bf*>(r + 16);
  return __builtin_shufflevector(lo, hi, 0, 1, 2, 3, 4, 5, 6, 7,
                                         8, 9, 10, 11, 12, 13, 14, 15);
}

static __device__ __forceinline__ v8f wmma_bf16(v16bf a, v16bf b, v8f c) {
  return __builtin_amdgcn_wmma_f32_16x16x32_bf16(false, a, false, b,
                                                 (short)0, c, false, false);
}

// ---- Tensor Data Mover: 2-D tile load, bf16 (data_size=2B) -----------------
// Descriptor per CDNA5 ISA. Groups 2/3 zero (2-D).
// padcfg ORs pad_enable/pad_interval/pad_amount into group1 word 0.
// 6-arg builtin form (clang-23 / therock-10.0 headers).
static __device__ __forceinline__ void tdm_load_2d(
    const void* gaddr, void* lds, unsigned td0, unsigned td1, unsigned tile0,
    unsigned tile1, unsigned stride0, unsigned padcfg) {
  unsigned long long ga = (unsigned long long)(uintptr_t)gaddr;
  u32x4 g0;
  g0[0] = 1u;                                    // count=1, user mode
  g0[1] = (unsigned)(uintptr_t)lds;              // LDS offset (addr[31:0])
  g0[2] = (unsigned)ga;                          // global_addr[31:0]
  g0[3] = (unsigned)(ga >> 32) | 0x80000000u;    // addr[56:32] | type=2
  i32x8 g1;
  g1[0] = (int)(0x00010000u | padcfg);           // data_size=2B (+pad cfg)
  g1[1] = (int)((td0 & 0xFFFFu) << 16);          // tensor_dim0[15:0]
  g1[2] = (int)((td0 >> 16) | ((td1 & 0xFFFFu) << 16));
  g1[3] = (int)((td1 >> 16) | (tile0 << 16));    // tile_dim0
  g1[4] = (int)tile1;                            // tile_dim1 (tile_dim2=0)
  g1[5] = (int)stride0;                          // tensor_dim0_stride[31:0]
  g1[6] = 0;
  g1[7] = 0;
  i32x4 z4 = {0, 0, 0, 0};
  i32x8 z8 = {0, 0, 0, 0, 0, 0, 0, 0};
  __builtin_amdgcn_tensor_load_to_lds(g0, g1, z4, z4, z8, 0);
}

// ---- prep kernels ----------------------------------------------------------
__global__ void cvt_bf16_kernel(const float* __restrict__ src,
                                bf16_t* __restrict__ dst, int n) {
  for (int i = blockIdx.x * blockDim.x + threadIdx.x; i < n;
       i += gridDim.x * blockDim.x)
    dst[i] = (bf16_t)src[i];
}

__global__ void zero32_kernel(unsigned* __restrict__ p, int n) {
  for (int i = blockIdx.x * blockDim.x + threadIdx.x; i < n;
       i += gridDim.x * blockDim.x)
    p[i] = 0u;
}

__global__ void bias_cat_kernel(const float* __restrict__ qb,
                                const float* __restrict__ vb,
                                float* __restrict__ out) {
  int c = blockIdx.x * blockDim.x + threadIdx.x;
  if (c < 3 * DIM)
    out[c] = (c < DIM) ? qb[c] : (c < 2 * DIM) ? 0.0f : vb[c - 2 * DIM];
}

// bias_pre[h][q][k] = rpb_table[rel_index[q*577+k]][h]
__global__ void gather_bias_kernel(const float* __restrict__ rpb,
                                   const int* __restrict__ rel,
                                   float* __restrict__ bp) {
  for (int i = blockIdx.x * blockDim.x + threadIdx.x; i < NN;
       i += gridDim.x * blockDim.x) {
    int idx = rel[i];
#pragma unroll
    for (int h = 0; h < NHEADS; ++h)
      bp[(size_t)h * NN + i] = rpb[idx * NHEADS + h];
  }
}

// ---- WMMA GEMM:  out[m][c] = sum_k A[m][k] * W[c][k]  (+ bias, epilogue) ---
// TDM-fed, double-buffered LDS. Wave0 drives the DMA; TENSORcnt gates buffers.
// EPI 0: scatter into scaled-Q / K / V bf16 buffers [B, h, NPAD, HDIM]
// EPI 1: f32 output [MROWS][DIM] + proj bias
template <int EPI>
__global__ __launch_bounds__(256) void gemm_kernel(
    const bf16_t* __restrict__ A, const bf16_t* __restrict__ W,
    const float* __restrict__ bias, bf16_t* __restrict__ qb,
    bf16_t* __restrict__ kb, bf16_t* __restrict__ vb,
    float* __restrict__ outf) {
  __shared__ __align__(16) bf16_t As[2][128][32];
  __shared__ __align__(16) bf16_t Bs[2][128][32];
  const int m0 = blockIdx.y * 128;
  const int c0 = blockIdx.x * 128;
  const int t = threadIdx.x;
  const int wave = t >> 5;
  const int wm = wave >> 2;  // 0..1 -> 64 rows each
  const int wn = wave & 3;   // 0..3 -> 32 cols each
  const unsigned mrem = (unsigned)(MROWS - m0);  // OOB rows -> zero via TDM

  v8f acc[4][2];
#pragma unroll
  for (int mf = 0; mf < 4; ++mf)
#pragma unroll
    for (int nf = 0; nf < 2; ++nf) acc[mf][nf] = (v8f){};

  if (wave == 0) {
    tdm_load_2d(A + (size_t)m0 * DIM, &As[0][0][0], 32, mrem, 32, 128, DIM, 0);
    tdm_load_2d(W + (size_t)c0 * DIM, &Bs[0][0][0], 32, 128, 32, 128, DIM, 0);
  }

  for (int ks = 0; ks < DIM / 32; ++ks) {
    const int buf = ks & 1;
    if (wave == 0) {
      if (ks + 1 < DIM / 32) {
        const int k1 = (ks + 1) * 32;
        tdm_load_2d(A + (size_t)m0 * DIM + k1, &As[buf ^ 1][0][0], 32, mrem,
                    32, 128, DIM, 0);
        tdm_load_2d(W + (size_t)c0 * DIM + k1, &Bs[buf ^ 1][0][0], 32, 128,
                    32, 128, DIM, 0);
        __builtin_amdgcn_s_wait_tensorcnt(2);  // current buffer's pair done
      } else {
        __builtin_amdgcn_s_wait_tensorcnt(0);
      }
    }
    __syncthreads();  // release current buffer

    v16bf bf0 = load_frag_b(&Bs[buf][wn * 32][0], 32);
    v16bf bf1 = load_frag_b(&Bs[buf][wn * 32 + 16][0], 32);
#pragma unroll
    for (int mf = 0; mf < 4; ++mf) {
      v16bf af = load_frag_a(&As[buf][wm * 64 + mf * 16][0], 32);
      acc[mf][0] = wmma_bf16(af, bf0, acc[mf][0]);
      acc[mf][1] = wmma_bf16(af, bf1, acc[mf][1]);
    }
    __syncthreads();  // all reads done before this buffer is DMA-overwritten
  }

  const int l = lane_id();
  const int colL = l & 15, halfL = l >> 4;
#pragma unroll
  for (int mf = 0; mf < 4; ++mf)
#pragma unroll
    for (int nf = 0; nf < 2; ++nf)
#pragma unroll
      for (int r = 0; r < 8; ++r) {
        int m = m0 + wm * 64 + mf * 16 + r + halfL * 8;
        int c = c0 + wn * 32 + nf * 16 + colL;
        if (m >= MROWS) continue;
        float v = acc[mf][nf][r] + bias[c];
        if (EPI == 0) {
          int which = c / DIM, rr = c % DIM;
          int h = rr / HDIM, dd = rr % HDIM;
          int b = m / NTOK, n = m % NTOK;
          size_t addr = (((size_t)(b * NHEADS + h)) * NPAD + n) * HDIM + dd;
          bf16_t o = (bf16_t)(which == 0 ? v * 0.125f : v);  // q pre-scaled
          (which == 0 ? qb : which == 1 ? kb : vb)[addr] = o;
        } else {
          outf[(size_t)m * DIM + c] = v;
        }
      }
}

// ---- flash attention per (b, h, 64-query tile) -----------------------------
__global__ __launch_bounds__(128) void flash_kernel(
    const bf16_t* __restrict__ qbuf, const bf16_t* __restrict__ kbuf,
    const bf16_t* __restrict__ vbuf, const float* __restrict__ bias_pre,
    bf16_t* __restrict__ attn_out) {
  __shared__ __align__(16) bf16_t Kt[64][72];      // [key][d], TDM-padded rows
  __shared__ __align__(16) bf16_t Vt[64][72];      // [d][key] (+pad)
  __shared__ __align__(16) bf16_t Ps[4][16][64];   // per-wave P scratch

  const int qtile = blockIdx.x, h = blockIdx.y, b = blockIdx.z;
  const int t = threadIdx.x, w = t >> 5, l = t & 31;
  const int colL = l & 15, halfL = l >> 4;
  const size_t bh = (size_t)(b * NHEADS + h);
  // TDM LDS padding: rows of 64 bf16 = 32 DWORDs (interval code 4),
  // insert 4 DWORDs (amount code 3) -> 72-element row pitch.
  const unsigned PADCFG = (1u << 20) | (4u << 22) | (3u << 25);

  // Q fragments stay in registers for the whole kernel (already scaled).
  const bf16_t* qp = qbuf + (bh * NPAD + qtile * 64 + w * 16) * HDIM;
  v16bf aq0 = load_frag_a(qp, HDIM);
  v16bf aq1 = load_frag_a(qp + 32, HDIM);

  float m8[8], l8[8];
  v8f accO[4];
#pragma unroll
  for (int r = 0; r < 8; ++r) { m8[r] = -3.0e38f; l8[r] = 0.0f; }
#pragma unroll
  for (int j = 0; j < 4; ++j) accO[j] = (v8f){};

  for (int kt = 0; kt < NPAD / 64; ++kt) {
    const int kb0 = kt * 64;
    if (w == 0)  // K tile via TDM, padded into bank-conflict-free pitch
      tdm_load_2d(kbuf + (bh * NPAD + kb0) * HDIM, &Kt[0][0], 64, 64, 64, 64,
                  HDIM, PADCFG);
    // V tile: manual transposed staging (TDM cannot transpose)
#pragma unroll
    for (int i = 0; i < 4; ++i) {
      int g = t + i * 128;             // 0..511
      int row = g >> 3, grp = g & 7;   // 64 keys x 8 chunks of 8 bf16
      v8bf vv = *reinterpret_cast<const v8bf*>(
          vbuf + (bh * NPAD + kb0 + row) * HDIM + grp * 8);
#pragma unroll
      for (int e = 0; e < 8; ++e) Vt[grp * 8 + e][row] = vv[e];  // transpose
    }
    if (w == 0) __builtin_amdgcn_s_wait_tensorcnt(0);
    __syncthreads();

    // S = Q * K^T  (16 x 64 per wave)
    v8f s[4];
#pragma unroll
    for (int nf = 0; nf < 4; ++nf) {
      v16bf b0 = load_frag_b(&Kt[nf * 16][0], 72);
      v16bf b1 = load_frag_b(&Kt[nf * 16][32], 72);
      v8f z = (v8f){};
      z = wmma_bf16(aq0, b0, z);
      s[nf] = wmma_bf16(aq1, b1, z);
    }

    // relative-position bias + key-padding mask
#pragma unroll
    for (int nf = 0; nf < 4; ++nf)
#pragma unroll
      for (int r = 0; r < 8; ++r) {
        int qg = qtile * 64 + w * 16 + r + halfL * 8;
        int kg = kb0 + nf * 16 + colL;
        float sv = s[nf][r];
        if (kg < NTOK) {
          if (qg < NTOK) sv += bias_pre[((size_t)h * NTOK + qg) * NTOK + kg];
        } else {
          sv = -1.0e30f;
        }
        s[nf][r] = sv;
      }

    // online softmax (rows live across 16-lane halves in the C layout)
    float mn8[8], rs8[8], al8[8];
#pragma unroll
    for (int r = 0; r < 8; ++r) {
      float mv = s[0][r];
      mv = fmaxf(mv, s[1][r]); mv = fmaxf(mv, s[2][r]); mv = fmaxf(mv, s[3][r]);
#pragma unroll
      for (int msk = 1; msk < 16; msk <<= 1)
        mv = fmaxf(mv, __shfl_xor(mv, msk, 32));
      mn8[r] = mv;
    }
#pragma unroll
    for (int r = 0; r < 8; ++r) {
      float mr = fmaxf(m8[r], mn8[r]);
      al8[r] = __expf(m8[r] - mr);
      m8[r] = mr;
      rs8[r] = 0.0f;
    }
#pragma unroll
    for (int nf = 0; nf < 4; ++nf)
#pragma unroll
      for (int r = 0; r < 8; ++r) {
        float p = __expf(s[nf][r] - m8[r]);
        s[nf][r] = p;
        rs8[r] += p;
      }
#pragma unroll
    for (int r = 0; r < 8; ++r) {
#pragma unroll
      for (int msk = 1; msk < 16; msk <<= 1)
        rs8[r] += __shfl_xor(rs8[r], msk, 32);
      l8[r] = l8[r] * al8[r] + rs8[r];
    }
#pragma unroll
    for (int j = 0; j < 4; ++j)
#pragma unroll
      for (int r = 0; r < 8; ++r) accO[j][r] *= al8[r];

    // P: C-layout -> A-layout via per-wave LDS round trip
#pragma unroll
    for (int nf = 0; nf < 4; ++nf)
#pragma unroll
      for (int r = 0; r < 8; ++r)
        Ps[w][r + halfL * 8][nf * 16 + colL] = (bf16_t)s[nf][r];
    v16bf ap0 = load_frag_a(&Ps[w][0][0], 64);
    v16bf ap1 = load_frag_a(&Ps[w][0][0] + 32, 64);

    // O += P * V
#pragma unroll
    for (int j = 0; j < 4; ++j) {
      v16bf b0 = load_frag_b(&Vt[j * 16][0], 72);
      v16bf b1 = load_frag_b(&Vt[j * 16][32], 72);
      accO[j] = wmma_bf16(ap0, b0, accO[j]);
      accO[j] = wmma_bf16(ap1, b1, accO[j]);
    }
    __syncthreads();
  }

  float inv8[8];
#pragma unroll
  for (int r = 0; r < 8; ++r) inv8[r] = (l8[r] > 0.0f) ? 1.0f / l8[r] : 0.0f;
#pragma unroll
  for (int j = 0; j < 4; ++j)
#pragma unroll
    for (int r = 0; r < 8; ++r) {
      int n = qtile * 64 + w * 16 + r + halfL * 8;
      if (n < NTOK)
        attn_out[((size_t)b * NTOK + n) * DIM + h * HDIM + j * 16 + colL] =
            (bf16_t)(accO[j][r] * inv8[r]);
    }
}

// ---- launcher --------------------------------------------------------------
extern "C" void kernel_launch(void* const* d_in, const int* in_sizes, int n_in,
                              void* d_out, int out_size, void* d_ws,
                              size_t ws_size, hipStream_t stream) {
  (void)in_sizes; (void)n_in; (void)out_size; (void)ws_size;
  const float* x      = (const float*)d_in[0];
  const float* qkv_w  = (const float*)d_in[1];
  const float* q_bias = (const float*)d_in[2];
  const float* v_bias = (const float*)d_in[3];
  const float* rpb    = (const float*)d_in[4];
  const float* proj_w = (const float*)d_in[5];
  const float* proj_b = (const float*)d_in[6];
  const int*   rel    = (const int*)d_in[7];
  float* out = (float*)d_out;

  char* wsp = (char*)d_ws;
  auto take = [&](size_t bytes) -> char* {
    char* p = wsp;
    wsp += (bytes + 255) & ~(size_t)255;
    return p;
  };
  bf16_t* xb   = (bf16_t*)take((size_t)MROWS * DIM * 2);
  bf16_t* qwb  = (bf16_t*)take((size_t)3 * DIM * DIM * 2);
  bf16_t* pwb  = (bf16_t*)take((size_t)DIM * DIM * 2);
  float*  qkvb = (float*)take((size_t)3 * DIM * 4);
  float*  bpre = (float*)take((size_t)NHEADS * NN * 4);
  const size_t qkvsz = (size_t)BATCH * NHEADS * NPAD * HDIM;
  bf16_t* qb  = (bf16_t*)take(qkvsz * 2);
  bf16_t* kb  = (bf16_t*)take(qkvsz * 2);
  bf16_t* vb  = (bf16_t*)take(qkvsz * 2);
  bf16_t* aob = (bf16_t*)take((size_t)MROWS * DIM * 2);

  cvt_bf16_kernel<<<2048, 256, 0, stream>>>(x, xb, MROWS * DIM);
  cvt_bf16_kernel<<<1024, 256, 0, stream>>>(qkv_w, qwb, 3 * DIM * DIM);
  cvt_bf16_kernel<<<512, 256, 0, stream>>>(proj_w, pwb, DIM * DIM);
  bias_cat_kernel<<<9, 256, 0, stream>>>(q_bias, v_bias, qkvb);
  gather_bias_kernel<<<1024, 256, 0, stream>>>(rpb, rel, bpre);
  zero32_kernel<<<2048, 256, 0, stream>>>((unsigned*)qb, (int)(qkvsz / 2));
  zero32_kernel<<<2048, 256, 0, stream>>>((unsigned*)kb, (int)(qkvsz / 2));
  zero32_kernel<<<2048, 256, 0, stream>>>((unsigned*)vb, (int)(qkvsz / 2));

  gemm_kernel<0><<<dim3(3 * DIM / 128, (MROWS + 127) / 128), 256, 0, stream>>>(
      xb, qwb, qkvb, qb, kb, vb, nullptr);

  flash_kernel<<<dim3(NPAD / 64, NHEADS, BATCH), 128, 0, stream>>>(
      qb, kb, vb, bpre, aob);

  gemm_kernel<1><<<dim3(DIM / 128, (MROWS + 127) / 128), 256, 0, stream>>>(
      aob, pwb, proj_b, nullptr, nullptr, nullptr, out);
}
